// ScaledDotProductAttention_27530740367593
// MI455X (gfx1250) — compile-verified, matrix operand order checked by reading, not run on hardware
//
#include <hip/hip_runtime.h>

#define B_ 2
#define H_ 16
#define L_ 2048
#define D_ 64

typedef __attribute__((ext_vector_type(16))) __bf16 v16bf;
typedef __attribute__((ext_vector_type(8)))  float  v8f;
typedef __attribute__((ext_vector_type(4)))  float  fvec4;

// Build a 16x32 bf16 A/B fragment from a row-major f32 row pointer.
// gfx1250 16-bit A/B layout: lane l (row/col = l&15, half = l>>4) holds
// element e at k = (e&7) + 8*half + 16*(e>>3) (+ 32*frag for the d-offset).
__device__ __forceinline__ v16bf load_frag_f32row(const float* __restrict__ rowp,
                                                  int half, int frag) {
  const float* base = rowp + 8 * half + 32 * frag;
  fvec4 a = *(const fvec4*)(base + 0);
  fvec4 b = *(const fvec4*)(base + 4);
  fvec4 c = *(const fvec4*)(base + 16);
  fvec4 d = *(const fvec4*)(base + 20);
  v16bf o;
  o[0]  = (__bf16)a[0]; o[1]  = (__bf16)a[1]; o[2]  = (__bf16)a[2]; o[3]  = (__bf16)a[3];
  o[4]  = (__bf16)b[0]; o[5]  = (__bf16)b[1]; o[6]  = (__bf16)b[2]; o[7]  = (__bf16)b[3];
  o[8]  = (__bf16)c[0]; o[9]  = (__bf16)c[1]; o[10] = (__bf16)c[2]; o[11] = (__bf16)c[3];
  o[12] = (__bf16)d[0]; o[13] = (__bf16)d[1]; o[14] = (__bf16)d[2]; o[15] = (__bf16)d[3];
  return o;
}

// ---------------- Pass 1: S = mask/bias(QK^T/8), raw scores -> scratch,
// per-row softmax stats (m, l) -> d_ws. Lane-private online stats; one
// cross-lane merge at the end (no per-tile bpermute storm). ----------------
__global__ __launch_bounds__(256) void attn_pass1_scores(
    const float* __restrict__ Q, const float* __restrict__ K,
    const int* __restrict__ mask, const float* __restrict__ bias,
    float* __restrict__ score, float* __restrict__ ws_m, float* __restrict__ ws_l) {
  const int lane = threadIdx.x & 31;
  const int lm   = lane & 15;
  const int half = lane >> 4;
  const int wave = blockIdx.x * (blockDim.x >> 5) + (threadIdx.x >> 5);
  const int QT = L_ / 16;
  const int idx = wave / QT;           // 0..B*H-1 (schedule index)
  const int q0  = (wave % QT) * 16;
  const int b   = idx & 1;             // pair b=0/b=1 of same head adjacent
  const int h   = idx >> 1;            // -> per-head bias slab reused in L2
  const int bh  = b * H_ + h;          // tensor index

  // Q tile A-fragments (lane holds row q0+lm), reused across all key tiles
  const float* qrow = Q + ((size_t)(bh * L_ + q0 + lm)) * D_;
  const v16bf aq0 = load_frag_f32row(qrow, half, 0);
  const v16bf aq1 = load_frag_f32row(qrow, half, 1);

  float mrun[8], lrun[8];
#pragma unroll
  for (int r = 0; r < 8; ++r) { mrun[r] = -3.0e38f; lrun[r] = 0.0f; }

  const int*   mbase = mask  + (size_t)b  * L_ * L_;
  const float* bbase = bias  + (size_t)h  * L_ * L_;
  float*       sbase = score + (size_t)bh * L_ * L_;

  for (int kt = 0; kt < L_ / 16; ++kt) {
    const int k0 = kt * 16;
    const float* krow = K + ((size_t)(bh * L_ + k0 + lm)) * D_;
    if (kt + 1 < L_ / 16) __builtin_prefetch(krow + 16 * D_, 0, 1);

    // B fragment: lane holds key column k0+lm (row of K == column of K^T)
    const v16bf bk0 = load_frag_f32row(krow, half, 0);
    const v16bf bk1 = load_frag_f32row(krow, half, 1);

    v8f c = {};
    c = __builtin_amdgcn_wmma_f32_16x16x32_bf16(false, aq0, false, bk0, (short)0, c, false, false);
    c = __builtin_amdgcn_wmma_f32_16x16x32_bf16(false, aq1, false, bk1, (short)0, c, false, false);

    // C layout: lane -> N = k0+lm, VGPR r -> M = q0 + r + 8*half
    const int n = k0 + lm;
#pragma unroll
    for (int r = 0; r < 8; ++r) {
      const int m = q0 + r + 8 * half;
      float s = c[r] * 0.125f;                                  // 1/sqrt(64)
      const int mk = mbase[(size_t)m * L_ + n];
      s = (mk == 0) ? -10000.0f : s;
      s += bbase[(size_t)m * L_ + n];                           // RT: L2-resident slab
      __builtin_nontemporal_store(s, &sbase[(size_t)m * L_ + n]); // raw score scratch
      // lane-private online softmax over this lane's column subset (1 exp/elt)
      const float d = s - mrun[r];
      const float e = __expf(-__builtin_fabsf(d));
      const bool gt = d > 0.0f;
      lrun[r] = gt ? (lrun[r] * e + 1.0f) : (lrun[r] + e);
      mrun[r] = gt ? s : mrun[r];
    }
  }

  // merge lane-private (m,l) across the 16 lanes that share each row
#pragma unroll
  for (int r = 0; r < 8; ++r) {
    float m = mrun[r], l = lrun[r];
#pragma unroll
    for (int off = 1; off <= 8; off <<= 1) {
      const float mo = __shfl_xor(m, off);
      const float lo = __shfl_xor(l, off);
      const float mn = fmaxf(m, mo);
      l = l * __expf(m - mn) + lo * __expf(mo - mn);
      m = mn;
    }
    if (lm == 0) {
      const int row = q0 + r + 8 * half;
      ws_m[(size_t)bh * L_ + row] = m;
      ws_l[(size_t)bh * L_ + row] = l;
    }
  }
}

// ---------------- Pass 2: probs = exp(s-m)/l (written as final output),
// out = probs @ V via WMMA; one shared V tile per block. ----------------
#define W2 4  // waves per block (all 4 share the same (b,h))
__global__ __launch_bounds__(128) void attn_pass2_pv(
    const float* __restrict__ V, float* __restrict__ score,
    const float* __restrict__ ws_m, const float* __restrict__ ws_l,
    float* __restrict__ out) {
  __shared__ __align__(16) float vlds[32 * 68];  // 32 keys x 64 d, stride 68

  const int lane = threadIdx.x & 31;
  const int lm   = lane & 15;
  const int half = lane >> 4;
  const int w    = threadIdx.x >> 5;
  const int wave = blockIdx.x * W2 + w;
  const int QT = L_ / 16;
  const int idx = wave / QT;
  const int q0  = (wave % QT) * 16;
  const int bh  = (idx & 1) * H_ + (idx >> 1);

  const float mrow = ws_m[(size_t)bh * L_ + q0 + lm];
  const float rl   = 1.0f / ws_l[(size_t)bh * L_ + q0 + lm];

  float*       srow = score + ((size_t)(bh * L_ + q0 + lm)) * L_;
  const float* vb   = V + (size_t)bh * L_ * D_;

  v8f acc[4] = {v8f{}, v8f{}, v8f{}, v8f{}};

  for (int ks = 0; ks < L_; ks += 32) {
    // --- stage V[ks..ks+32) x 64 into LDS cooperatively (all 4 waves) ---
#pragma unroll
    for (int it = 0; it < 4; ++it) {
      const int cidx = it * 128 + threadIdx.x;  // 512 float4 chunks
      const int row  = cidx >> 4;               // 0..31
      const int ch   = cidx & 15;               // float4 within row
      fvec4 v4 = *(const fvec4*)(vb + (size_t)(ks + row) * D_ + ch * 4);
      *(fvec4*)(vlds + row * 68 + ch * 4) = v4;
    }

    // --- raw scores for 32 keys of this row, in A-fragment order ---
    float* sp = srow + ks + 8 * half;
    fvec4 s0 = __builtin_nontemporal_load((const fvec4*)(sp + 0));
    fvec4 s1 = __builtin_nontemporal_load((const fvec4*)(sp + 4));
    fvec4 s2 = __builtin_nontemporal_load((const fvec4*)(sp + 16));
    fvec4 s3 = __builtin_nontemporal_load((const fvec4*)(sp + 20));
    if (ks + 32 < L_) __builtin_prefetch(sp + 32, 0, 0);
#pragma unroll
    for (int i = 0; i < 4; ++i) {
      s0[i] = __expf(s0[i] - mrow) * rl;
      s1[i] = __expf(s1[i] - mrow) * rl;
      s2[i] = __expf(s2[i] - mrow) * rl;
      s3[i] = __expf(s3[i] - mrow) * rl;
    }
    __builtin_nontemporal_store(s0, (fvec4*)(sp + 0));   // final probs output
    __builtin_nontemporal_store(s1, (fvec4*)(sp + 4));
    __builtin_nontemporal_store(s2, (fvec4*)(sp + 16));
    __builtin_nontemporal_store(s3, (fvec4*)(sp + 20));

    v16bf ap;
    ap[0]  = (__bf16)s0[0]; ap[1]  = (__bf16)s0[1]; ap[2]  = (__bf16)s0[2]; ap[3]  = (__bf16)s0[3];
    ap[4]  = (__bf16)s1[0]; ap[5]  = (__bf16)s1[1]; ap[6]  = (__bf16)s1[2]; ap[7]  = (__bf16)s1[3];
    ap[8]  = (__bf16)s2[0]; ap[9]  = (__bf16)s2[1]; ap[10] = (__bf16)s2[2]; ap[11] = (__bf16)s2[3];
    ap[12] = (__bf16)s3[0]; ap[13] = (__bf16)s3[1]; ap[14] = (__bf16)s3[2]; ap[15] = (__bf16)s3[3];

    __syncthreads();

    // --- B fragments: lane holds V column nt*16+lm, K striped ---
#pragma unroll
    for (int nt = 0; nt < 4; ++nt) {
      v16bf bv;
#pragma unroll
      for (int e = 0; e < 16; ++e) {
        const int k = (e & 7) + 8 * half + 16 * (e >> 3);
        bv[e] = (__bf16)vlds[k * 68 + nt * 16 + lm];
      }
      acc[nt] = __builtin_amdgcn_wmma_f32_16x16x32_bf16(false, ap, false, bv, (short)0,
                                                        acc[nt], false, false);
    }
    __syncthreads();
  }

  // write out: C layout lane -> d = nt*16+lm, VGPR r -> m = q0 + r + 8*half
  float* ob = out + ((size_t)bh * L_ + q0) * D_;
#pragma unroll
  for (int nt = 0; nt < 4; ++nt) {
#pragma unroll
    for (int r = 0; r < 8; ++r) {
      const int m = r + 8 * half;
      ob[(size_t)m * D_ + nt * 16 + lm] = acc[nt][r];
    }
  }
}

extern "C" void kernel_launch(void* const* d_in, const int* in_sizes, int n_in,
                              void* d_out, int out_size, void* d_ws, size_t ws_size,
                              hipStream_t stream) {
  const float* Q    = (const float*)d_in[0];
  const float* K    = (const float*)d_in[1];
  const float* V    = (const float*)d_in[2];
  const int*   mask = (const int*)d_in[3];
  const float* bias = (const float*)d_in[4];
  // d_in[5] = last_layer_flag == 0 -> (out, probs) path (matches out_size)

  float* out   = (float*)d_out;
  float* score = out + (size_t)B_ * H_ * L_ * D_;   // second tuple element
  float* ws_m  = (float*)d_ws;                      // [B*H*L]
  float* ws_l  = ws_m + (size_t)B_ * H_ * L_;       // [B*H*L]

  const int waves = B_ * H_ * (L_ / 16);  // 4096 query tiles
  attn_pass1_scores<<<waves / 8, 256, 0, stream>>>(Q, K, mask, bias, score, ws_m, ws_l);
  attn_pass2_pv<<<waves / W2, W2 * 32, 0, stream>>>(V, score, ws_m, ws_l, out);
}
